// EglGATConv_56599079026978
// MI455X (gfx1250) — matrix operand order repaired and checked by reading, not compiled, not run on hardware
//
#include <hip/hip_runtime.h>

#define N_NODES 100000
#define N_EDGES 3200000
#define NHEAD   4
#define HDIM    16
#define F_IN    256
#define F_OUT   64
#define NEG_SLOPE 0.2f

typedef __attribute__((ext_vector_type(16))) __bf16 v16bf;
typedef __attribute__((ext_vector_type(8)))  float  v8f;

__device__ __forceinline__ float lrelu(float x) { return x > 0.f ? x : NEG_SLOPE * x; }

__device__ __forceinline__ void pack8(v16bf& v, int base, float4 p, float4 q) {
  v[base + 0] = (__bf16)p.x; v[base + 1] = (__bf16)p.y;
  v[base + 2] = (__bf16)p.z; v[base + 3] = (__bf16)p.w;
  v[base + 4] = (__bf16)q.x; v[base + 5] = (__bf16)q.y;
  v[base + 6] = (__bf16)q.z; v[base + 7] = (__bf16)q.w;
}

// ---------------------------------------------------------------------------
// Kernel 0: init emax = -inf, denom = 0
// ---------------------------------------------------------------------------
__global__ void gat_init_kernel(float* __restrict__ emax, float* __restrict__ denom) {
  int t = blockIdx.x * blockDim.x + threadIdx.x;
  if (t < N_NODES * NHEAD) {
    emax[t]  = -__builtin_inff();
    denom[t] = 0.f;
  }
}

// ---------------------------------------------------------------------------
// Kernel 1: WMMA projection. One wave per 16-node tile. Computes
//   feat_src = feat @ W_fc^T  -> workspace
//   resval   = feat @ W_res^T -> d_out (output init)
// bf16 A/B, f32 accumulate via v_wmma_f32_16x16x32_bf16.
// ---------------------------------------------------------------------------
__global__ __launch_bounds__(256) void gat_proj_kernel(
    const float* __restrict__ feat, const float* __restrict__ Wfc,
    const float* __restrict__ Wres, float* __restrict__ feat_src,
    float* __restrict__ out)
{
  const int wave = threadIdx.x >> 5;
  const int lane = threadIdx.x & 31;
  const int tile = blockIdx.x * 8 + wave;
  if (tile * 16 >= N_NODES) return;

  const int m0   = tile * 16;
  const int half = lane >> 4;   // 0: lanes 0-15, 1: lanes 16-31
  const int l15  = lane & 15;

  const float* arow = feat + (size_t)(m0 + l15) * F_IN;

  v8f zero = {0.f, 0.f, 0.f, 0.f, 0.f, 0.f, 0.f, 0.f};
  v8f acc_fc[4], acc_res[4];
#pragma unroll
  for (int i = 0; i < 4; ++i) { acc_fc[i] = zero; acc_res[i] = zero; }

#pragma unroll
  for (int kc = 0; kc < F_IN; kc += 32) {
    // A tile (16x32 bf16): lanes 0-15 hold K kc+0..15,
    //                      lanes 16-31 hold K kc+16..31
    const int kbA = kc + half * 16;
    float4 a0 = *(const float4*)(arow + kbA);
    float4 a1 = *(const float4*)(arow + kbA + 4);
    float4 a2 = *(const float4*)(arow + kbA + 8);
    float4 a3 = *(const float4*)(arow + kbA + 12);
    v16bf a;
    pack8(a, 0, a0, a1);
    pack8(a, 8, a2, a3);

    // B tile (32x16 bf16): lane holds column n0+l15, K kbB..kbB+15
    const int kbB = kc + half * 16;
#pragma unroll
    for (int nt = 0; nt < 4; ++nt) {
      const int n0 = nt * 16;
      {
        const float* wr = Wfc + (size_t)(n0 + l15) * F_IN + kbB;
        float4 b0 = *(const float4*)(wr);
        float4 b1 = *(const float4*)(wr + 4);
        float4 b2 = *(const float4*)(wr + 8);
        float4 b3 = *(const float4*)(wr + 12);
        v16bf b;
        pack8(b, 0, b0, b1);
        pack8(b, 8, b2, b3);
        acc_fc[nt] = __builtin_amdgcn_wmma_f32_16x16x32_bf16(
            false, a, false, b, (short)0, acc_fc[nt], false, false);
      }
      {
        const float* wr = Wres + (size_t)(n0 + l15) * F_IN + kbB;
        float4 b0 = *(const float4*)(wr);
        float4 b1 = *(const float4*)(wr + 4);
        float4 b2 = *(const float4*)(wr + 8);
        float4 b3 = *(const float4*)(wr + 12);
        v16bf b;
        pack8(b, 0, b0, b1);
        pack8(b, 8, b2, b3);
        acc_res[nt] = __builtin_amdgcn_wmma_f32_16x16x32_bf16(
            false, a, false, b, (short)0, acc_res[nt], false, false);
      }
    }
  }

  // C/D layout: VGPR r -> row m0 + r + 8*half, col n0 + l15
#pragma unroll
  for (int nt = 0; nt < 4; ++nt) {
#pragma unroll
    for (int r = 0; r < 8; ++r) {
      const int row = m0 + r + 8 * half;
      const int col = nt * 16 + l15;
      feat_src[(size_t)row * F_OUT + col] = acc_fc[nt][r];
      out[(size_t)row * F_OUT + col]      = acc_res[nt][r];
    }
  }
}

// ---------------------------------------------------------------------------
// Kernel 2: per (node, head) attention logits el/er
// ---------------------------------------------------------------------------
__global__ void gat_elr_kernel(const float* __restrict__ fs,
                               const float* __restrict__ attn_l,
                               const float* __restrict__ attn_r,
                               float* __restrict__ el, float* __restrict__ er)
{
  int t = blockIdx.x * blockDim.x + threadIdx.x;
  if (t >= N_NODES * NHEAD) return;
  const int n = t >> 2, h = t & 3;
  const float* row = fs + (size_t)n * F_OUT + h * HDIM;
  const float* al  = attn_l + h * HDIM;
  const float* ar  = attn_r + h * HDIM;
  float sl = 0.f, sr = 0.f;
#pragma unroll
  for (int i = 0; i < HDIM; ++i) {
    float v = row[i];
    sl += v * al[i];
    sr += v * ar[i];
  }
  el[t] = sl;
  er[t] = sr;
}

// ---------------------------------------------------------------------------
// Kernel 3: segment max over dst (float atomicMax via int trick)
// ---------------------------------------------------------------------------
__global__ void gat_emax_kernel(const int* __restrict__ src, const int* __restrict__ dst,
                                const float* __restrict__ el, const float* __restrict__ er,
                                float* __restrict__ emax)
{
  int t = blockIdx.x * blockDim.x + threadIdx.x;
  if (t >= N_EDGES * NHEAD) return;
  const int e = t >> 2, h = t & 3;
  const int s = src[e], d = dst[e];
  const float v = lrelu(el[s * NHEAD + h] + er[d * NHEAD + h]);
  float* a = emax + d * NHEAD + h;
  if (v >= 0.f) atomicMax((int*)a, __float_as_int(v));
  else          atomicMin((unsigned int*)a, (unsigned int)__float_as_int(v));
}

// ---------------------------------------------------------------------------
// Kernel 4: denom = segment_sum(exp(e - emax[dst]))
// ---------------------------------------------------------------------------
__global__ void gat_esum_kernel(const int* __restrict__ src, const int* __restrict__ dst,
                                const float* __restrict__ el, const float* __restrict__ er,
                                const float* __restrict__ emax, float* __restrict__ denom)
{
  int t = blockIdx.x * blockDim.x + threadIdx.x;
  if (t >= N_EDGES * NHEAD) return;
  const int e = t >> 2, h = t & 3;
  const int s = src[e], d = dst[e];
  const float v  = lrelu(el[s * NHEAD + h] + er[d * NHEAD + h]);
  const float ee = __expf(v - emax[d * NHEAD + h]);
  atomicAdd(denom + d * NHEAD + h, ee);
}

// ---------------------------------------------------------------------------
// Kernel 5: weighted scatter-add. 64 threads per edge (one per out channel);
// alpha recomputed from L2-hot per-node arrays.
// ---------------------------------------------------------------------------
__global__ __launch_bounds__(256) void gat_agg_kernel(
    const int* __restrict__ src, const int* __restrict__ dst,
    const float* __restrict__ el, const float* __restrict__ er,
    const float* __restrict__ emax, const float* __restrict__ denom,
    const float* __restrict__ fs, float* __restrict__ out)
{
  long long t = (long long)blockIdx.x * 256 + threadIdx.x;
  if (t >= (long long)N_EDGES * F_OUT) return;
  const int e = (int)(t >> 6);
  const int c = (int)t & 63;      // c = h*16 + d
  const int h = c >> 4;
  const int s = src[e], d = dst[e];
  const float v     = lrelu(el[s * NHEAD + h] + er[d * NHEAD + h]);
  const float ee    = __expf(v - emax[d * NHEAD + h]);
  const float alpha = ee / fmaxf(denom[d * NHEAD + h], 1e-20f);
  atomicAdd(out + (size_t)d * F_OUT + c, alpha * fs[(size_t)s * F_OUT + c]);
}

// ---------------------------------------------------------------------------
extern "C" void kernel_launch(void* const* d_in, const int* in_sizes, int n_in,
                              void* d_out, int out_size, void* d_ws, size_t ws_size,
                              hipStream_t stream) {
  const float* feat   = (const float*)d_in[0];
  const float* W_fc   = (const float*)d_in[1];
  const float* attn_l = (const float*)d_in[2];
  const float* attn_r = (const float*)d_in[3];
  const float* W_res  = (const float*)d_in[4];
  const int*   src    = (const int*)d_in[5];
  const int*   dst    = (const int*)d_in[6];
  float* out = (float*)d_out;

  // workspace layout (floats): feat_src | el | er | emax | denom  (32 MB total)
  float* ws       = (float*)d_ws;
  float* feat_src = ws;
  float* el       = feat_src + (size_t)N_NODES * F_OUT;   // 6,400,000
  float* er       = el + (size_t)N_NODES * NHEAD;         // +400,000
  float* emax     = er + (size_t)N_NODES * NHEAD;
  float* denom    = emax + (size_t)N_NODES * NHEAD;

  const int nh = N_NODES * NHEAD;          // 400,000
  const int eh = N_EDGES * NHEAD;          // 12,800,000

  gat_init_kernel<<<(nh + 255) / 256, 256, 0, stream>>>(emax, denom);

  const int ntiles = N_NODES / 16;         // 6250
  gat_proj_kernel<<<(ntiles + 7) / 8, 256, 0, stream>>>(feat, W_fc, W_res, feat_src, out);

  gat_elr_kernel<<<(nh + 255) / 256, 256, 0, stream>>>(feat_src, attn_l, attn_r, el, er);

  gat_emax_kernel<<<(eh + 255) / 256, 256, 0, stream>>>(src, dst, el, er, emax);
  gat_esum_kernel<<<(eh + 255) / 256, 256, 0, stream>>>(src, dst, el, er, emax, denom);

  const long long total = (long long)N_EDGES * F_OUT;    // 204,800,000
  const int agg_blocks = (int)((total + 255) / 256);     // 800,000
  gat_agg_kernel<<<agg_blocks, 256, 0, stream>>>(src, dst, el, er, emax, denom,
                                                 feat_src, out);
}
